// BiLSTM_CRF_NER_10574209482925
// MI455X (gfx1250) — matrix operand order repaired and checked by reading, not compiled, not run on hardware
//
#include <hip/hip_runtime.h>
#include <hip/hip_bf16.h>

typedef __attribute__((ext_vector_type(16))) _Float16 v16h;
typedef __attribute__((ext_vector_type(8)))  _Float16 v8h;
typedef __attribute__((ext_vector_type(8)))  float    v8f;
typedef __attribute__((ext_vector_type(4)))  float    v4f;

#define CRF_B 128
#define CRF_T 512
#define CRF_K 64

#define LOG2E 1.4426950408889634f
#define LN2   0.6931471805599453f

// ---------------------------------------------------------------------------
// Gold-path score: score[b] = sum_t mask * (emit[b,t,tag] + trans[prev,tag])
// ---------------------------------------------------------------------------
__global__ void crf_gold_kernel(const float* __restrict__ emit,
                                const float* __restrict__ trans,
                                const int* __restrict__ tags,
                                const int* __restrict__ lens,
                                float* __restrict__ gold) {
  int b = blockIdx.x * blockDim.x + threadIdx.x;
  if (b >= CRF_B) return;
  const int len = lens[b];
  int prev = tags[b * CRF_T];
  float s = emit[((size_t)b * CRF_T) * CRF_K + prev];   // t=0, len>=1 always
  for (int t = 1; t < len; ++t) {                       // mask == (t < len)
    const int tg = tags[b * CRF_T + t];
    s += emit[((size_t)b * CRF_T + t) * CRF_K + tg] + trans[prev * CRF_K + tg];
    prev = tg;
  }
  gold[b] = s;
}

// ---------------------------------------------------------------------------
// Forward algorithm (log-partition) via WMMA, base-2 throughout.
// One wave per 16-row batch tile; grid = 8 waves. d kept in registers
// (owner layout: lane owns row nl, a 32-col half). LDS is used only for
// the WMMA A-matrix staging and the C-layout -> owner-layout transpose.
// emit is loaded in owner layout (contiguous -> b128) and added during the
// branch-free masked blend.
// ---------------------------------------------------------------------------
__global__ __launch_bounds__(32)
void crf_fwd_kernel(const float* __restrict__ emit,
                    const float* __restrict__ trans,
                    const int* __restrict__ lens,
                    const float* __restrict__ gold,
                    float* __restrict__ out) {
  __shared__ __align__(16) _Float16 pls[16][CRF_K];     // exp2 staging (A matrix)
  __shared__ __align__(16) float    ndls[16][CRF_K];    // new d staging (C layout)
  __shared__ __align__(16) _Float16 Els[CRF_K][CRF_K];  // exp(trans)

  const int l   = threadIdx.x;       // 0..31
  const int b0  = blockIdx.x * 16;   // batch tile base
  const int nl  = l & 15;
  const int hi  = l >> 4;            // half-wave id
  const int ro  = nl;                // owned row
  const int c0  = hi * 32;           // owned column block
  const int hi8 = hi * 8;            // C-layout M offset
  const int kb0 = hi * 8;            // A-layout K offset

  // E = exp(trans) -> LDS (BOS row / EOS col at -1e4 underflow to exactly 0)
  for (int idx = l; idx < CRF_K * CRF_K; idx += 32)
    Els[idx >> 6][idx & 63] =
        (_Float16)__builtin_amdgcn_exp2f(trans[idx] * LOG2E);
  __syncthreads();

  // Constant B-matrix fragments (16-bit B layout: lanes 0-15 hold K=0..7,16..23,
  // lanes 16-31 hold K=8..15,24..31, 2 K per VGPR).
  v16h ef[4][2];
  #pragma unroll
  for (int n = 0; n < 4; ++n)
    #pragma unroll
    for (int c = 0; c < 2; ++c)
      #pragma unroll
      for (int e = 0; e < 16; ++e)
        ef[n][c][e] = Els[c * 32 + hi * 8 + (e & 7) + (e >> 3) * 16][n * 16 + nl];

  // d state in registers, owner layout: lane owns (row ro, cols c0..c0+31)
  const int len_ro = lens[b0 + ro];   // this lane's row length
  const int lenmax = lens[b0];        // lengths sorted descending -> tile max
  const float* erow = emit + ((size_t)(b0 + ro) * CRF_T) * CRF_K + c0;

  float dreg[32];
  #pragma unroll
  for (int q = 0; q < 8; ++q) {
    const v4f f = *(const v4f*)(erow + 4 * q);          // t = 0
    #pragma unroll
    for (int x = 0; x < 4; ++x) dreg[4 * q + x] = f[x];
  }

  for (int t = 1; t < lenmax; ++t) {
    // ---- issue this step's emit loads first (owner layout, contiguous) ----
    float ev[32];
    {
      const float* ep = erow + (size_t)t * CRF_K;
      #pragma unroll
      for (int q = 0; q < 8; ++q) {
        const v4f f = *(const v4f*)(ep + 4 * q);
        #pragma unroll
        for (int x = 0; x < 4; ++x) ev[4 * q + x] = f[x];
      }
    }

    // ---- phase 1: row max + p = exp2((d-m)*log2e), packed f16 stores ----
    float m = -__builtin_inff();
    #pragma unroll
    for (int j = 0; j < 32; ++j) m = fmaxf(m, dreg[j]);
    m = fmaxf(m, __shfl_xor(m, 16, 32));
    {
      v8h pk[4];
      #pragma unroll
      for (int j = 0; j < 32; ++j)
        pk[j >> 3][j & 7] =
            (_Float16)__builtin_amdgcn_exp2f((dreg[j] - m) * LOG2E);
      #pragma unroll
      for (int q = 0; q < 4; ++q)
        *(v8h*)&pls[ro][c0 + q * 8] = pk[q];
    }
    __syncthreads();

    // ---- A fragments (16-bit A layout, 16B LDS loads) ----
    v16h a0, a1;
    {
      const v8h lo0 = *(const v8h*)&pls[nl][kb0];
      const v8h hh0 = *(const v8h*)&pls[nl][kb0 + 16];
      const v8h lo1 = *(const v8h*)&pls[nl][kb0 + 32];
      const v8h hh1 = *(const v8h*)&pls[nl][kb0 + 48];
      #pragma unroll
      for (int e = 0; e < 8; ++e) {
        a0[e] = lo0[e]; a0[e + 8] = hh0[e];
        a1[e] = lo1[e]; a1[e + 8] = hh1[e];
      }
    }

    // row maxes for C-layout rows, via lane broadcast
    float rm[8];
    #pragma unroll
    for (int r = 0; r < 8; ++r) rm[r] = __shfl(m, r + hi8, 32);

    // ---- WMMA + epilogue: stage rm + ln2*log2(acc) to LDS (C layout) ----
    #pragma unroll
    for (int n = 0; n < 4; ++n) {
      v8f acc = {};
      acc = __builtin_amdgcn_wmma_f32_16x16x32_f16(false, a0, false, ef[n][0],
                                                   (short)0, acc, false, false);
      acc = __builtin_amdgcn_wmma_f32_16x16x32_f16(false, a1, false, ef[n][1],
                                                   (short)0, acc, false, false);
      #pragma unroll
      for (int r = 0; r < 8; ++r)
        ndls[r + hi8][n * 16 + nl] =
            rm[r] + LN2 * __builtin_amdgcn_logf(acc[r]);
    }
    __syncthreads();

    // ---- branch-free masked blend (+ emit) back into the register copy ----
    const bool upd = (t < len_ro);       // per-lane uniform -> v_cndmask
    #pragma unroll
    for (int j = 0; j < 32; ++j) {
      const float nv = ndls[ro][c0 + j] + ev[j];
      dreg[j] = upd ? nv : dreg[j];
    }
  }

  // ---- final: log_z[b] = lse_j(d[b,j]);  out = log_z - gold ----
  {
    float m = -__builtin_inff();
    #pragma unroll
    for (int j = 0; j < 32; ++j) m = fmaxf(m, dreg[j]);
    m = fmaxf(m, __shfl_xor(m, 16, 32));
    float s = 0.f;
    #pragma unroll
    for (int j = 0; j < 32; ++j)
      s += __builtin_amdgcn_exp2f((dreg[j] - m) * LOG2E);
    s += __shfl_xor(s, 16, 32);
    if (hi == 0)
      out[b0 + ro] = (m + LN2 * __builtin_amdgcn_logf(s)) - gold[b0 + ro];
  }
}

// ---------------------------------------------------------------------------
extern "C" void kernel_launch(void* const* d_in, const int* in_sizes, int n_in,
                              void* d_out, int out_size, void* d_ws, size_t ws_size,
                              hipStream_t stream) {
  (void)in_sizes; (void)n_in; (void)out_size; (void)ws_size;
  const float* emit  = (const float*)d_in[0];   // (B,T,K) f32
  const float* trans = (const float*)d_in[1];   // (K,K)   f32
  const int*   tags  = (const int*)d_in[2];     // (B,T)   int
  const int*   lens  = (const int*)d_in[3];     // (B,)    int, sorted desc
  float* out  = (float*)d_out;                  // (B,)    f32
  float* gold = (float*)d_ws;                   // B floats of scratch

  crf_gold_kernel<<<1, 128, 0, stream>>>(emit, trans, tags, lens, gold);
  crf_fwd_kernel<<<CRF_B / 16, 32, 0, stream>>>(emit, trans, lens, gold, out);
}
